// Timeline_77094662963559
// MI455X (gfx1250) — compile-verified, matrix operand order checked by reading, not run on hardware
//
#include <hip/hip_runtime.h>

typedef __attribute__((ext_vector_type(16))) __bf16 v16bf;
typedef __attribute__((ext_vector_type(8)))  float  v8f;

// ---------------------------------------------------------------------------
// WMMA helpers (CDNA5 wave32, 16x16x32 bf16 -> f32 accumulate)
// A-operand layout (16-bit, 16x32): lane = (laneHalf<<4)|m ; element e maps to
//   K = (e&7) + ((e>>3)<<4) + laneHalf*8.  B-operand (32x16, B[k][n]=W[n][k]
// row-major) uses the identical per-lane pattern with n = lane&15, so one
// loader serves A tiles, transposed-B tiles (Q@K^T) and weight-B tiles.
// ---------------------------------------------------------------------------
__device__ __forceinline__ v16bf loadA(const __bf16* base, int pitch,
                                       int r0, int c0, int lane) {
  int m  = lane & 15;
  int hh = (lane >> 4) & 1;
  const __bf16* p = base + (r0 + m) * pitch + c0 + (hh << 3);
  v16bf a;
#pragma unroll
  for (int e = 0; e < 8; ++e) { a[e] = p[e]; a[e + 8] = p[e + 16]; }
  return a;
}

__device__ __forceinline__ v8f wmma_bf16(v16bf a, v16bf b, v8f c) {
  return __builtin_amdgcn_wmma_f32_16x16x32_bf16(
      false, a, false, b, (short)0, c, false, false);
}

__device__ __forceinline__ float sigm(float x) { return 1.f / (1.f + __expf(-x)); }

typedef union { __bf16 h[2]; unsigned int u; } bfpack;

// ---------------------------------------------------------------------------
// fp32 -> bf16 weight conversion
// ---------------------------------------------------------------------------
__global__ void cvt_kernel(const float* __restrict__ src,
                           __bf16* __restrict__ dst, int n) {
  int i = blockIdx.x * blockDim.x + threadIdx.x;
  if (i < n) dst[i] = (__bf16)src[i];
}

// ---------------------------------------------------------------------------
// Per-visit attention: one block per (b,v) visit. 256 threads = 8 waves.
//   emb[64x256] gathered to bf16 LDS; Q,K = emb @ W^T via WMMA;
//   S = softmax((QK^T - maskR - maskC)/sqrt(128)); w = (gate*keep)^T S;
//   vv = w @ emb  -> written time-major [v][b][256] as bf16.
// ---------------------------------------------------------------------------
#define EP 264
#define QP 136
#define SP 68
#define ATTN_SMEM 70656

__global__ __launch_bounds__(256) void attn_kernel(
    const int* __restrict__ codes, const float* __restrict__ mask,
    const float* __restrict__ tvals, const float* __restrict__ emb_table,
    const float* __restrict__ decay, const float* __restrict__ initial,
    const __bf16* __restrict__ wq_bf, const __bf16* __restrict__ wk_bf,
    __bf16* __restrict__ vv_bf) {
  extern __shared__ char smem[];
  __bf16* sEmb  = (__bf16*)(smem);            // [64][EP]   33792 B
  __bf16* sQ    = (__bf16*)(smem + 33792);    // [64][QP]   17408 B
  __bf16* sK    = (__bf16*)(smem + 51200);    // [64][QP]   17408 B
  float*  sS    = (float*) (smem + 33792);    // [64][SP]   aliases sQ
  float*  sMask = (float*) (smem + 68608);    // [64]
  float*  sG    = (float*) (smem + 68864);    // [64]
  float*  sW    = (float*) (smem + 69120);    // [64]
  float*  sRed  = (float*) (smem + 69376);    // [4][64]
  int*    sCode = (int*)   (smem + 70400);    // [64]

  const int n    = blockIdx.x;           // n = b*50 + v
  const int tid  = threadIdx.x;
  const int lane = tid & 31;
  const int wv   = tid >> 5;             // wave id 0..7

  if (tid < 64) {
    int   code = codes[n * 64 + tid];
    float m    = mask[n * 64 + tid];
    sCode[tid] = code;
    sMask[tid] = m;
    float keep = (m - 1e20f) * (-1e-20f);                 // 1 real, 0 pad
    float z    = decay[code] * tvals[n] + initial[code];
    sG[tid]    = keep * sigm(z);                          // gate * keep
  }
  __syncthreads();

  // gather embedding rows as packed bf16 pairs (8B global read per lane/iter)
  {
    unsigned int* sEmbU = (unsigned int*)sEmb;
    for (int p = 0; p < 32; ++p) {
      int pi = tid + p * 256;           // pair index, 0..8191
      int r = pi >> 7, cp = pi & 127;
      int code = sCode[r];
      const float* src = emb_table + code * 256 + cp * 2;
      bfpack pk;
      pk.h[0] = (__bf16)src[0];
      pk.h[1] = (__bf16)src[1];
      sEmbU[r * (EP / 2) + cp] = pk.u;
    }
  }
  __syncthreads();

  // ---- Q, K projections: wave wv owns column tile nt = wv (cols wv*16..)
  v8f qa[4], ka[4];
#pragma unroll
  for (int mt = 0; mt < 4; ++mt)
#pragma unroll
    for (int i = 0; i < 8; ++i) { qa[mt][i] = 0.f; ka[mt][i] = 0.f; }

  for (int kk = 0; kk < 8; ++kk) {             // K = 256 = 8 * 32
    v16bf bq = loadA(wq_bf, 256, wv * 16, kk * 32, lane);
    v16bf bk = loadA(wk_bf, 256, wv * 16, kk * 32, lane);
#pragma unroll
    for (int mt = 0; mt < 4; ++mt) {
      v16bf a = loadA(sEmb, EP, mt * 16, kk * 32, lane);
      qa[mt] = wmma_bf16(a, bq, qa[mt]);
      ka[mt] = wmma_bf16(a, bk, ka[mt]);
    }
  }
  {
    int h = lane >> 4, nn = lane & 15;
#pragma unroll
    for (int mt = 0; mt < 4; ++mt)
#pragma unroll
      for (int i = 0; i < 8; ++i) {
        int row = mt * 16 + i + 8 * h;
        sQ[row * QP + wv * 16 + nn] = (__bf16)qa[mt][i];
        sK[row * QP + wv * 16 + nn] = (__bf16)ka[mt][i];
      }
  }
  __syncthreads();

  // ---- S = Q @ K^T  (K-tile loaded as transposed-B == A-style load of K)
  v8f sacc[2];
#pragma unroll
  for (int ti = 0; ti < 2; ++ti)
#pragma unroll
    for (int i = 0; i < 8; ++i) sacc[ti][i] = 0.f;

#pragma unroll
  for (int ti = 0; ti < 2; ++ti) {
    int t = wv * 2 + ti, mt = t >> 2, nt = t & 3;
    for (int kk = 0; kk < 4; ++kk) {           // K = 128 = 4 * 32
      v16bf a = loadA(sQ, QP, mt * 16, kk * 32, lane);
      v16bf b = loadA(sK, QP, nt * 16, kk * 32, lane);
      sacc[ti] = wmma_bf16(a, b, sacc[ti]);
    }
  }
  __syncthreads();   // all Q/K reads done before sS aliases sQ
  const float isc = 0.08838834764831845f;      // 1/sqrt(128)
#pragma unroll
  for (int ti = 0; ti < 2; ++ti) {
    int t = wv * 2 + ti, mt = t >> 2, nt = t & 3;
    int h = lane >> 4, nn = lane & 15;
    int col = nt * 16 + nn;
    float mcol = sMask[col];
#pragma unroll
    for (int i = 0; i < 8; ++i) {
      int row = mt * 16 + i + 8 * h;
      sS[row * SP + col] = (sacc[ti][i] - sMask[row] - mcol) * isc;
    }
  }
  __syncthreads();

  // ---- softmax rows (4 threads per row, fp32)
  {
    int r = tid >> 2, sub = tid & 3;
    float buf[16]; float mx = -3.4e38f;
#pragma unroll
    for (int j = 0; j < 16; ++j) {
      buf[j] = sS[r * SP + sub * 16 + j];
      mx = fmaxf(mx, buf[j]);
    }
    mx = fmaxf(mx, __shfl_xor(mx, 1, 4));
    mx = fmaxf(mx, __shfl_xor(mx, 2, 4));
    float sum = 0.f;
#pragma unroll
    for (int j = 0; j < 16; ++j) { buf[j] = __expf(buf[j] - mx); sum += buf[j]; }
    sum += __shfl_xor(sum, 1, 4);
    sum += __shfl_xor(sum, 2, 4);
    float inv = 1.f / sum;
#pragma unroll
    for (int j = 0; j < 16; ++j) sS[r * SP + sub * 16 + j] = buf[j] * inv;
  }
  __syncthreads();

  // ---- w[d] = sum_c g[c] * S[c][d]
  {
    int part = tid >> 6, col = tid & 63;
    float acc = 0.f;
    for (int r = part * 16; r < part * 16 + 16; ++r)
      acc += sG[r] * sS[r * SP + col];
    sRed[part * 64 + col] = acc;
  }
  __syncthreads();
  if (tid < 64)
    sW[tid] = sRed[tid] + sRed[64 + tid] + sRed[128 + tid] + sRed[192 + tid];
  __syncthreads();

  // ---- vv[e] = sum_d w[d] * emb[d][e]   (time-major output)
  {
    float acc = 0.f;
#pragma unroll 4
    for (int d = 0; d < 64; ++d)
      acc += sW[d] * (float)sEmb[d * EP + tid];
    int b = n / 50, v = n % 50;
    vv_bf[(v * 32 + b) * 256 + tid] = (__bf16)acc;
  }
}

// ---------------------------------------------------------------------------
// Input projections: out = X @ W^T + b1 + b2.  One wave per 16x16 tile, K=256.
// tiled!=0: write C-layout tiles (tile*256 + lane*8 + i) so the recurrent
// kernel can vector-load accumulator fragments.  tiled==0: plain [M][1024].
// ---------------------------------------------------------------------------
__global__ __launch_bounds__(32) void pregemm_kernel(
    const __bf16* __restrict__ X, const __bf16* __restrict__ W,
    const float* __restrict__ b1, const float* __restrict__ b2,
    float* __restrict__ out, int tiled) {
  int nt = blockIdx.x, mt = blockIdx.y;
  int lane = threadIdx.x;
  v8f acc;
#pragma unroll
  for (int i = 0; i < 8; ++i) acc[i] = 0.f;
  for (int kk = 0; kk < 8; ++kk) {
    v16bf a = loadA(X, 256, mt * 16, kk * 32, lane);
    v16bf b = loadA(W, 256, nt * 16, kk * 32, lane);
    acc = wmma_bf16(a, b, acc);
  }
  int h = lane >> 4, nn = lane & 15;
  int j = nt * 16 + nn;
  float bias = b1[j] + b2[j];
  if (tiled) {
    v8f o;
#pragma unroll
    for (int i = 0; i < 8; ++i) o[i] = acc[i] + bias;
    *(v8f*)(out + ((size_t)(mt * 64 + nt)) * 256 + lane * 8) = o;
  } else {
#pragma unroll
    for (int i = 0; i < 8; ++i) {
      int m = mt * 16 + i + 8 * h;
      out[m * 1024 + j] = acc[i] + bias;
    }
  }
}

// ---------------------------------------------------------------------------
// Forward LSTM recurrence: single persistent block, 16 waves.
// Wave w owns gate columns [w*16, w*16+16) of all 4 gates, both batch tiles;
// cell state c stays in registers (C-layout), h round-trips via bf16 LDS.
// preF is C-layout-tiled: acc fragments load as one v8f per tile.
// ---------------------------------------------------------------------------
#define HP 264
__global__ __launch_bounds__(512) void lstm_fwd_kernel(
    const float* __restrict__ preF, const __bf16* __restrict__ whh_bf,
    float* __restrict__ hf_out) {
  __shared__ __bf16 sH[32 * HP];
  int tid = threadIdx.x, lane = tid & 31, wv = tid >> 5;   // wv 0..15
  for (int i = tid; i < 32 * HP; i += 512) sH[i] = (__bf16)0.f;

  v8f c[2];
#pragma unroll
  for (int mt = 0; mt < 2; ++mt)
#pragma unroll
    for (int i = 0; i < 8; ++i) c[mt][i] = 0.f;

  int h = lane >> 4, nn = lane & 15;
  int jn = wv * 16 + nn;

  for (int t = 0; t < 50; ++t) {
    __syncthreads();                       // sH(t) visible
    if (t + 1 < 50)
      __builtin_prefetch(preF + ((size_t)((t + 1) * 128 + wv)) * 256 + lane * 8,
                         0, 1);

    v8f acc[2][4];
#pragma unroll
    for (int mt = 0; mt < 2; ++mt)
#pragma unroll
      for (int q = 0; q < 4; ++q)
        acc[mt][q] = *(const v8f*)(
            preF + ((size_t)((t * 2 + mt) * 64 + q * 16 + wv)) * 256 + lane * 8);

    for (int kk = 0; kk < 8; ++kk) {       // K = 256
      v16bf a0 = loadA(sH, HP, 0,  kk * 32, lane);
      v16bf a1 = loadA(sH, HP, 16, kk * 32, lane);
#pragma unroll
      for (int q = 0; q < 4; ++q) {
        v16bf b = loadA(whh_bf, 256, q * 256 + wv * 16, kk * 32, lane);
        acc[0][q] = wmma_bf16(a0, b, acc[0][q]);
        acc[1][q] = wmma_bf16(a1, b, acc[1][q]);
      }
    }
    __syncthreads();                       // all sH reads done
#pragma unroll
    for (int mt = 0; mt < 2; ++mt)
#pragma unroll
      for (int i = 0; i < 8; ++i) {
        float iv = sigm(acc[mt][0][i]);
        float fv = sigm(acc[mt][1][i]);
        float gv = tanhf(acc[mt][2][i]);
        float ov = sigm(acc[mt][3][i]);
        float cv = fv * c[mt][i] + iv * gv;
        c[mt][i] = cv;
        float hv = ov * tanhf(cv);
        int m = mt * 16 + i + 8 * h;
        sH[m * HP + jn] = (__bf16)hv;
        if (t == 49) hf_out[m * 256 + jn] = hv;
      }
  }
}

// ---------------------------------------------------------------------------
// Head: backward LSTM single step (h0=c0=0 => pure elementwise on preB),
// then logits = [hf|hb] @ W_out^T + b_out, row softmax.
// ---------------------------------------------------------------------------
__global__ __launch_bounds__(256) void head_kernel(
    const float* __restrict__ preB, const float* __restrict__ hfF,
    const float* __restrict__ W_out, const float* __restrict__ b_out,
    float* __restrict__ out) {
  __shared__ float sHB[32 * 256];
  __shared__ float sLog[320];
  int tid = threadIdx.x;
  for (int idx = tid; idx < 8192; idx += 256) {
    int b = idx >> 8, k = idx & 255;
    const float* p = preB + b * 1024 + k;
    float iv = sigm(p[0]);
    float gv = tanhf(p[512]);
    float ov = sigm(p[768]);
    float cv = iv * gv;                      // f-gate * 0 drops
    sHB[idx] = ov * tanhf(cv);
  }
  __syncthreads();
  for (int idx = tid; idx < 320; idx += 256) {
    int b = idx / 10, l = idx % 10;
    float acc = b_out[l];
    const float* wrow = W_out + l * 512;
    const float* hf = hfF + b * 256;
    const float* hb = sHB + b * 256;
    for (int k = 0; k < 256; ++k)
      acc += hf[k] * wrow[k] + hb[k] * wrow[256 + k];
    sLog[idx] = acc;
  }
  __syncthreads();
  if (tid < 32) {
    float mx = -3.4e38f;
    for (int l = 0; l < 10; ++l) mx = fmaxf(mx, sLog[tid * 10 + l]);
    float s = 0.f, e[10];
    for (int l = 0; l < 10; ++l) { e[l] = __expf(sLog[tid * 10 + l] - mx); s += e[l]; }
    float inv = 1.f / s;
    for (int l = 0; l < 10; ++l) out[tid * 10 + l] = e[l] * inv;
  }
}

// ---------------------------------------------------------------------------
extern "C" void kernel_launch(void* const* d_in, const int* in_sizes, int n_in,
                              void* d_out, int out_size, void* d_ws, size_t ws_size,
                              hipStream_t stream) {
  (void)in_sizes; (void)n_in; (void)out_size; (void)ws_size;
  const int*   codes   = (const int*)  d_in[0];
  const float* mask    = (const float*)d_in[1];
  const float* tvals   = (const float*)d_in[2];
  const float* emb     = (const float*)d_in[4];
  const float* WQ1     = (const float*)d_in[5];
  const float* WK1     = (const float*)d_in[6];
  const float* decay   = (const float*)d_in[7];
  const float* initial = (const float*)d_in[8];
  const float* w_ih_f  = (const float*)d_in[9];
  const float* w_hh_f  = (const float*)d_in[10];
  const float* b_ih_f  = (const float*)d_in[11];
  const float* b_hh_f  = (const float*)d_in[12];
  const float* w_ih_b  = (const float*)d_in[13];
  const float* b_ih_b  = (const float*)d_in[15];
  const float* b_hh_b  = (const float*)d_in[16];
  const float* W_out   = (const float*)d_in[17];
  const float* b_out   = (const float*)d_in[18];
  float* out = (float*)d_out;

  char* ws = (char*)d_ws;
  __bf16* wq_bf   = (__bf16*)(ws + 0);         //  65536 B
  __bf16* wk_bf   = (__bf16*)(ws + 65536);     //  65536 B
  __bf16* wihf_bf = (__bf16*)(ws + 131072);    // 524288 B
  __bf16* wihb_bf = (__bf16*)(ws + 655360);    // 524288 B
  __bf16* whhf_bf = (__bf16*)(ws + 1179648);   // 524288 B
  __bf16* vv_bf   = (__bf16*)(ws + 1703936);   // 819200 B  [50*32][256]
  float*  preF    = (float*) (ws + 2523136);   // 6553600 B tiled [100*64][256]
  float*  preB    = (float*) (ws + 9076736);   // 131072 B  [32][1024]
  float*  hfF     = (float*) (ws + 9207808);   //  32768 B  [32][256]

  cvt_kernel<<<128, 256, 0, stream>>>(WQ1, wq_bf, 32768);
  cvt_kernel<<<128, 256, 0, stream>>>(WK1, wk_bf, 32768);
  cvt_kernel<<<1024, 256, 0, stream>>>(w_ih_f, wihf_bf, 262144);
  cvt_kernel<<<1024, 256, 0, stream>>>(w_ih_b, wihb_bf, 262144);
  cvt_kernel<<<1024, 256, 0, stream>>>(w_hh_f, whhf_bf, 262144);

  attn_kernel<<<1600, 256, ATTN_SMEM, stream>>>(
      codes, mask, tvals, emb, decay, initial, wq_bf, wk_bf, vv_bf);

  pregemm_kernel<<<dim3(64, 100), 32, 0, stream>>>(
      vv_bf, wihf_bf, b_ih_f, b_hh_f, preF, 1);
  pregemm_kernel<<<dim3(64, 2), 32, 0, stream>>>(
      vv_bf + 1568 * 256, wihb_bf, b_ih_b, b_hh_b, preB, 0);

  lstm_fwd_kernel<<<1, 512, 0, stream>>>(preF, whhf_bf, hfF);
  head_kernel<<<1, 256, 0, stream>>>(preB, hfF, W_out, b_out, out);
}
